// PatchEncoder_22222160789764
// MI455X (gfx1250) — matrix operand order repaired and checked by reading, not compiled, not run on hardware
//
#include <hip/hip_runtime.h>
#include <hip/hip_bf16.h>

// ---------------------------------------------------------------------------
// PatchEncoder for MI455X (gfx1250, wave32, WMMA)
//   z-embed -> per-filter (relu(Z W1 + b1)) W2 + b2 -> masked softmax over S
//   -> weighted sum against z.  25.6 GFLOP, fully L2-resident => matrix-core
//   bound => f16 WMMA (f32 accumulate) via v_wmma_f32_16x16x32_f16.
// ---------------------------------------------------------------------------

typedef _Float16 v16h __attribute__((ext_vector_type(16)));
typedef _Float16 v8h  __attribute__((ext_vector_type(8)));
typedef float    v8f  __attribute__((ext_vector_type(8)));

#define D_TIME 63
#define D_IN   64
#define D_OUT  127
#define D_HID  128
#define NB     8
#define NP     32
#define S_LEN  48
#define NROWS  (NB * NP)          // 256 patches
#define NZ     (NROWS * S_LEN)    // 12288 event rows
#define NEGV   (-1e9f)
#define HSTR   72                 // padded f16 row stride in LDS (72*2B=144B, no bank conflicts)

union AU { v16h h16; v8h h8[2]; };

// ---------------------------------------------------------------------------
// Kernel 1: continuous-time embedding.  z[r][0..62] = time embed, z[r][63] = v.
// Writes f32 (for the final weighted sum) and f16 (WMMA A operand source).
// ---------------------------------------------------------------------------
__global__ __launch_bounds__(256)
void k_embed(const float* __restrict__ values, const float* __restrict__ times,
             const float* __restrict__ omega, const float* __restrict__ alpha,
             _Float16* __restrict__ zf16, float* __restrict__ zf32) {
    int g = blockIdx.x * 256 + threadIdx.x;
    if (g >= NZ * D_IN) return;
    int r = g >> 6;
    int i = g & 63;
    float t = times[r];
    float z;
    if (i == 0)             z = omega[0] * t + alpha[0];
    else if (i < D_TIME)    z = sinf(t * omega[i] + alpha[i]);
    else                    z = values[r];
    zf32[g] = z;
    zf16[g] = (_Float16)z;
}

// ---------------------------------------------------------------------------
// Kernel 2: W1/W2 (f32, [d][i][j]) -> f16 transposed [d][j][i] so a WMMA
// B-operand lane (lane = column j, 16 contiguous K values) is one 32B load.
// ---------------------------------------------------------------------------
__global__ __launch_bounds__(256)
void k_wt(const float* __restrict__ W1, const float* __restrict__ W2,
          _Float16* __restrict__ w1t, _Float16* __restrict__ w2t) {
    const int per = D_OUT * D_IN * D_IN;     // 520192
    int g = blockIdx.x * 256 + threadIdx.x;
    if (g >= 2 * per) return;
    const float* src = (g < per) ? W1 : W2;
    _Float16*    dst = (g < per) ? w1t : w2t;
    int e   = (g < per) ? g : g - per;
    int d   = e / (D_IN * D_IN);
    int rem = e - d * (D_IN * D_IN);
    int j   = rem >> 6;                      // output row  (= column of W)
    int i   = rem & 63;                      // k
    dst[d * (D_IN * D_IN) + j * D_IN + i] =
        (_Float16)src[d * (D_IN * D_IN) + i * D_IN + j];
}

// ---------------------------------------------------------------------------
// Kernel 3: mask-any channel (output column 127).
// (jax bool arrays are 1 byte/elem; treat mask as u8.)
// ---------------------------------------------------------------------------
__global__ void k_tail(const unsigned char* __restrict__ mask, float* __restrict__ out) {
    int n = threadIdx.x;
    if (n < NROWS) {
        unsigned any = 0;
        for (int s = 0; s < S_LEN; ++s) any |= mask[n * S_LEN + s];
        out[(size_t)n * D_HID + (D_HID - 1)] = any ? 1.0f : 0.0f;
    }
}

// ---------------------------------------------------------------------------
// Kernel 4: main per-(patch, filter) pipeline.
//   grid = (127 filters, 256 patches), block = 128 threads = 4 waves.
//   Wave w owns 16 output columns [16w, 16w+16).
//   GEMM1: A = z (global f16, contiguous K order: per lane two 16B
//          chunks at K {0..7,8..15}/{16..23,24..31}), B = w1t tile, K = 64.
//   relu + f16 -> LDS H (stride 72), barrier,
//   GEMM2: A = H from LDS (ds_load_b128 pairs), B = w2t, + bias -> LDS logits,
//   barrier, masked softmax over s fused with the Σ_s,j w·z reduction.
// ---------------------------------------------------------------------------
__global__ __launch_bounds__(128)
void k_main(const _Float16* __restrict__ zf16, const float* __restrict__ zf32,
            const _Float16* __restrict__ w1t,  const _Float16* __restrict__ w2t,
            const float* __restrict__ b1,      const float* __restrict__ b2,
            const unsigned char* __restrict__ mask, float* __restrict__ out) {
    const int d    = blockIdx.x;              // filter 0..126
    const int n    = blockIdx.y;              // patch  0..255
    const int tid  = threadIdx.x;
    const int wv   = tid >> 5;                // wave 0..3 (wave32)
    const int lane = tid & 31;
    const int lm   = lane & 15;
    const bool hi  = lane >= 16;
    const int col  = wv * 16 + lm;            // output column this lane owns

    __shared__ _Float16 Hs[S_LEN * HSTR];     // 6912 B
    __shared__ float    Ls[S_LEN * D_IN];     // 12288 B
    __shared__ unsigned msk[S_LEN];
    __shared__ float    hacc;

    if (tid < S_LEN) msk[tid] = mask[n * S_LEN + tid];
    if (tid == 0)    hacc = 0.0f;

    const float bias1 = b1[d * D_IN + col];
    const float bias2 = b2[d * D_IN + col];

    // ---------------- GEMM1: H = relu(Z @ W1[d] + b1[d]) ----------------
    v8f acc1[3] = {};
    {
        const _Float16* zb = zf16 + (size_t)n * (S_LEN * D_IN);
        const _Float16* wb = w1t + (size_t)d * (D_IN * D_IN) + (size_t)col * D_IN;
        #pragma unroll
        for (int ks = 0; ks < 2; ++ks) {
            v16h bm = *(const v16h*)(wb + ks * 32 + (hi ? 16 : 0));
            #pragma unroll
            for (int mt = 0; mt < 3; ++mt) {
                const _Float16* ap = zb + (mt * 16 + lm) * D_IN + ks * 32 + (hi ? 16 : 0);
                AU a;
                a.h8[0] = *(const v8h*)(ap);
                a.h8[1] = *(const v8h*)(ap + 8);
                acc1[mt] = __builtin_amdgcn_wmma_f32_16x16x32_f16(
                    false, a.h16, false, bm, (short)0, acc1[mt], false, false);
            }
        }
    }
    #pragma unroll
    for (int mt = 0; mt < 3; ++mt) {
        #pragma unroll
        for (int r = 0; r < 8; ++r) {
            int row = mt * 16 + (hi ? r + 8 : r);   // C/D layout: VGPR r = rows M=r / M=r+8
            float h = acc1[mt][r] + bias1;
            Hs[row * HSTR + col] = (_Float16)(h > 0.0f ? h : 0.0f);
        }
    }
    __syncthreads();

    // ---------------- GEMM2: logits = H @ W2[d] + b2[d] ----------------
    v8f acc2[3] = {};
    {
        const _Float16* wb = w2t + (size_t)d * (D_IN * D_IN) + (size_t)col * D_IN;
        #pragma unroll
        for (int ks = 0; ks < 2; ++ks) {
            v16h bm = *(const v16h*)(wb + ks * 32 + (hi ? 16 : 0));
            #pragma unroll
            for (int mt = 0; mt < 3; ++mt) {
                const _Float16* hp = &Hs[(mt * 16 + lm) * HSTR + ks * 32 + (hi ? 16 : 0)];
                AU a;
                a.h8[0] = *(const v8h*)(hp);
                a.h8[1] = *(const v8h*)(hp + 8);
                acc2[mt] = __builtin_amdgcn_wmma_f32_16x16x32_f16(
                    false, a.h16, false, bm, (short)0, acc2[mt], false, false);
            }
        }
    }
    #pragma unroll
    for (int mt = 0; mt < 3; ++mt) {
        #pragma unroll
        for (int r = 0; r < 8; ++r) {
            int row = mt * 16 + (hi ? r + 8 : r);
            Ls[row * D_IN + col] = acc2[mt][r] + bias2;
        }
    }
    __syncthreads();

    // -------- masked softmax over s (axis=1) fused with Σ_s w·z --------
    if (tid < D_IN) {
        const int j = tid;
        float mx = -3.0e38f;
        for (int s = 0; s < S_LEN; ++s) {
            float l = msk[s] ? Ls[s * D_IN + j] : NEGV;
            mx = fmaxf(mx, l);
        }
        float den = 0.0f, num = 0.0f;
        const float* zr = zf32 + (size_t)n * (S_LEN * D_IN) + j;
        for (int s = 0; s < S_LEN; ++s) {
            float l = msk[s] ? Ls[s * D_IN + j] : NEGV;
            float e = __expf(l - mx);
            den += e;
            num += e * zr[(size_t)s * D_IN];
        }
        atomicAdd(&hacc, num / den);          // ds_add_f32, 64 lanes -> one scalar
    }
    __syncthreads();
    if (tid == 0) out[(size_t)n * D_HID + d] = hacc;
}

// ---------------------------------------------------------------------------
// Workspace layout (bytes):
//   zf16 : [0,        1572864)   12288*64 f16
//   w1t  : [1572864,  2613248)   127*64*64 f16 (transposed)
//   w2t  : [2613248,  3653632)   127*64*64 f16 (transposed)
//   zf32 : [3653632,  6799360)   12288*64 f32
// ---------------------------------------------------------------------------
extern "C" void kernel_launch(void* const* d_in, const int* in_sizes, int n_in,
                              void* d_out, int out_size, void* d_ws, size_t ws_size,
                              hipStream_t stream) {
    (void)in_sizes; (void)n_in; (void)out_size; (void)ws_size;

    const float*         values = (const float*)d_in[0];
    const float*         times  = (const float*)d_in[1];
    const unsigned char* maskp  = (const unsigned char*)d_in[2];
    const float*         omega  = (const float*)d_in[3];
    const float*         alpha  = (const float*)d_in[4];
    const float*         W1     = (const float*)d_in[5];
    const float*         b1     = (const float*)d_in[6];
    const float*         W2     = (const float*)d_in[7];
    const float*         b2     = (const float*)d_in[8];
    float*               out    = (float*)d_out;

    char* ws = (char*)d_ws;
    _Float16* zf16 = (_Float16*)(ws);
    _Float16* w1t  = (_Float16*)(ws + 1572864);
    _Float16* w2t  = (_Float16*)(ws + 2613248);
    float*    zf32 = (float*)   (ws + 3653632);

    k_embed<<<(NZ * D_IN + 255) / 256, 256, 0, stream>>>(values, times, omega, alpha, zf16, zf32);
    k_wt  <<<(2 * D_OUT * D_IN * D_IN + 255) / 256, 256, 0, stream>>>(W1, W2, w1t, w2t);
    k_tail<<<1, 256, 0, stream>>>(maskp, out);
    k_main<<<dim3(D_OUT, NROWS), 128, 0, stream>>>(zf16, zf32, w1t, w2t, b1, b2, maskp, out);
}
